// GNNStack_11166914970396
// MI455X (gfx1250) — matrix operand order repaired and checked by reading, not compiled, not run on hardware
//
#include <hip/hip_runtime.h>
#include <math.h>

// ---------------------------------------------------------------------------
// GNNStack forward for MI455X (gfx1250, wave32, WMMA).
// All dense GEMMs use v_wmma_f32_16x16x32_f16 (f16 in, f32 accum).
// Edge aggregation uses f16 gathers + global_atomic_add_f32 (L2-resident).
// ---------------------------------------------------------------------------

#define HD 128          // hidden dim
#define LIN_IN 243      // (11-2)^2 * 3
#define KPAD 256        // padded K for the 243->128 GEMM

typedef _Float16 half4_t  __attribute__((ext_vector_type(4)));
typedef _Float16 half8_t  __attribute__((ext_vector_type(8)));
typedef _Float16 half16_t __attribute__((ext_vector_type(16)));
typedef float    float8_t __attribute__((ext_vector_type(8)));

// ---------------- WMMA fragment loaders ------------------------------------
// A: row-major f16 [M, ldk]. ISA 16-bit A 16x32 layout:
//   lanes 0-15  (row = lane): halves 0-7 = K k0..k0+7,  halves 8-15 = K k0+16..k0+23
//   lanes 16-31 (row = lane-16): same with +8 on K base.
__device__ __forceinline__ half16_t ldA(const _Float16* __restrict__ A,
                                        int row0, int k0, int ldk) {
  int lane = threadIdx.x & 31;
  const _Float16* p = A + (size_t)(row0 + (lane & 15)) * ldk
                        + k0 + ((lane & 16) ? 8 : 0);
  half8_t lo = *(const half8_t*)p;         // K = base .. base+7
  half8_t hi = *(const half8_t*)(p + 16);  // K = base+16 .. base+23
  half16_t r;
#pragma unroll
  for (int i = 0; i < 8; ++i) { r[i] = lo[i]; r[i + 8] = hi[i]; }
  return r;
}

// B: weights pre-transposed to [col][K] f16 (column's K contiguous).
// Layout (same K order as ldA):
//   lanes 0-15 (col = lane): halves 0-7 = K k0..k0+7, halves 8-15 = K k0+16..k0+23;
//   lanes 16-31: same with +8 on K base.
__device__ __forceinline__ half16_t ldB(const _Float16* __restrict__ Wt,
                                        int col0, int k0, int ldk) {
  int lane = threadIdx.x & 31;
  const _Float16* p = Wt + (size_t)(col0 + (lane & 15)) * ldk
                         + k0 + ((lane & 16) ? 8 : 0);
  half8_t lo = *(const half8_t*)p;
  half8_t hi = *(const half8_t*)(p + 16);
  half16_t r;
#pragma unroll
  for (int i = 0; i < 8; ++i) { r[i] = lo[i]; r[i + 8] = hi[i]; }
  return r;
}

// ---------------- weight prep ----------------------------------------------
// lin_w / lin_self_w [243,128] f32 -> f16 [128][256] transposed, zero padded.
__global__ void __launch_bounds__(256)
prep_padT(const float* __restrict__ w1, const float* __restrict__ w2,
          _Float16* __restrict__ t1, _Float16* __restrict__ t2) {
  int i = blockIdx.x * 256 + threadIdx.x;     // over 128*256
  if (i >= HD * KPAD) return;
  int c = i >> 8, k = i & 255;
  _Float16 a = (k < LIN_IN) ? (_Float16)w1[k * HD + c] : (_Float16)0.f;
  _Float16 b = (k < LIN_IN) ? (_Float16)w2[k * HD + c] : (_Float16)0.f;
  t1[i] = a; t2[i] = b;
}

// [128,128] f32 -> f16 [128][128] transposed.
__global__ void __launch_bounds__(256)
prep_T128(const float* __restrict__ w, _Float16* __restrict__ t) {
  int i = blockIdx.x * 256 + threadIdx.x;     // over 128*128
  if (i >= HD * HD) return;
  int c = i >> 7, k = i & 127;
  t[i] = (_Float16)w[k * HD + c];
}

// ---------------- conv (both branches) -> f16 padded rows -------------------
__global__ void __launch_bounds__(256)
conv_kernel(const float* __restrict__ x,
            const float* __restrict__ cw,  const float* __restrict__ cb,
            const float* __restrict__ csw, const float* __restrict__ csb,
            _Float16* __restrict__ cNb, _Float16* __restrict__ cSelf) {
  __shared__ float sx[363];
  __shared__ float sw[2][81];
  __shared__ float sb[2][3];
  int n = blockIdx.x, t = threadIdx.x;
  const float* xin = x + (size_t)n * 363;
  for (int i = t; i < 363; i += 256) sx[i] = xin[i];
  if (t < 81)                sw[0][t]       = cw[t];
  else if (t < 162)          sw[1][t - 81]  = csw[t - 81];
  else if (t < 165)          sb[0][t - 162] = cb[t - 162];
  else if (t < 168)          sb[1][t - 165] = csb[t - 165];
  __syncthreads();
  for (int o = t; o < 2 * LIN_IN; o += 256) {
    int br = (o >= LIN_IN);
    int rem = o - br * LIN_IN;
    int c = rem / 81, p = rem % 81, oy = p / 9, ox = p % 9;
    const float* wp = &sw[br][c * 27];
    float s = sb[br][c];
#pragma unroll
    for (int ci = 0; ci < 3; ++ci)
#pragma unroll
      for (int ky = 0; ky < 3; ++ky)
#pragma unroll
        for (int kx = 0; kx < 3; ++kx)
          s += sx[ci * 121 + (oy + ky) * 11 + (ox + kx)] * wp[ci * 9 + ky * 3 + kx];
    s = fmaxf(s, 0.f);
    (br ? cSelf : cNb)[(size_t)n * KPAD + rem] = (_Float16)s;
  }
  if (t >= LIN_IN && t < KPAD) {   // zero K padding 243..255
    cNb[(size_t)n * KPAD + t]   = (_Float16)0.f;
    cSelf[(size_t)n * KPAD + t] = (_Float16)0.f;
  }
}

// ---------------- GEMM [N,256] f16 @ [256,128] -> f32 + bias ----------------
__global__ void __launch_bounds__(256)
gemm_k256(const _Float16* __restrict__ A, const _Float16* __restrict__ Wt,
          const float* __restrict__ bias, float* __restrict__ C, int nTiles) {
  int tile = blockIdx.x * (blockDim.x >> 5) + (threadIdx.x >> 5);
  if (tile >= nTiles) return;
  int row0 = tile << 4;
  half16_t a[8];
#pragma unroll
  for (int k = 0; k < 8; ++k) a[k] = ldA(A, row0, k * 32, KPAD);
  int lane = threadIdx.x & 31;
  int cl = lane & 15, rOff = (lane & 16) ? 8 : 0;
#pragma unroll
  for (int nt = 0; nt < 8; ++nt) {
    float8_t acc = {};
#pragma unroll
    for (int k = 0; k < 8; ++k) {
      half16_t b = ldB(Wt, nt * 16, k * 32, KPAD);
      acc = __builtin_amdgcn_wmma_f32_16x16x32_f16(false, a[k], false, b,
                                                   (short)0, acc, false, false);
    }
    int col = nt * 16 + cl;
    float bv = bias[col];
#pragma unroll
    for (int r = 0; r < 8; ++r)
      C[(size_t)(row0 + rOff + r) * HD + col] = acc[r] + bv;
  }
}

// ------ fused [N,128]x[128,128] GEMM; DUAL: agg@lw + h@rw + bias (SAGE) -----
template <bool DUAL>
__global__ void __launch_bounds__(256)
gemm128(const _Float16* __restrict__ A1, const _Float16* __restrict__ A2,
        const _Float16* __restrict__ W1, const _Float16* __restrict__ W2,
        const float* __restrict__ bias, float* __restrict__ C, int nTiles) {
  int tile = blockIdx.x * (blockDim.x >> 5) + (threadIdx.x >> 5);
  if (tile >= nTiles) return;
  int row0 = tile << 4;
  half16_t a1[4], a2[4];
#pragma unroll
  for (int k = 0; k < 4; ++k) {
    a1[k] = ldA(A1, row0, k * 32, HD);
    if (DUAL) a2[k] = ldA(A2, row0, k * 32, HD);
  }
  int lane = threadIdx.x & 31;
  int cl = lane & 15, rOff = (lane & 16) ? 8 : 0;
#pragma unroll
  for (int nt = 0; nt < 8; ++nt) {
    float8_t acc = {};
#pragma unroll
    for (int k = 0; k < 4; ++k) {
      half16_t b = ldB(W1, nt * 16, k * 32, HD);
      acc = __builtin_amdgcn_wmma_f32_16x16x32_f16(false, a1[k], false, b,
                                                   (short)0, acc, false, false);
      if (DUAL) {
        half16_t b2 = ldB(W2, nt * 16, k * 32, HD);
        acc = __builtin_amdgcn_wmma_f32_16x16x32_f16(false, a2[k], false, b2,
                                                     (short)0, acc, false, false);
      }
    }
    int col = nt * 16 + cl;
    float bv = bias[col];
#pragma unroll
    for (int r = 0; r < 8; ++r)
      C[(size_t)(row0 + rOff + r) * HD + col] = acc[r] + bv;
  }
}

// ---------------- edge kernels ----------------------------------------------
__global__ void __launch_bounds__(256)
cnt_kernel(const int* __restrict__ dst, float* __restrict__ cnt, int E) {
  int e = blockIdx.x * 256 + threadIdx.x;
  if (e < E) atomicAdd(&cnt[dst[e]], 1.0f);
}

// layer-0 scatter: xn[src] (f32) -> h[dst], skip self loops. 1 wave = 1 edge.
__global__ void __launch_bounds__(256)
scatter_xn(const int* __restrict__ src, const int* __restrict__ dst,
           const float* __restrict__ xn, float* __restrict__ h, int E) {
  int tid = blockIdx.x * 256 + threadIdx.x;
  int e = tid >> 5;
  if (e >= E) return;
  int s = src[e], d = dst[e];
  if (s == d) return;
  int f = (tid & 31) * 4;
  float4 v = *(const float4*)(xn + (size_t)s * HD + f);
  float* hp = h + (size_t)d * HD + f;
  atomicAdd(hp + 0, v.x); atomicAdd(hp + 1, v.y);
  atomicAdd(hp + 2, v.z); atomicAdd(hp + 3, v.w);
}

// SAGE scatter: hPost[src] (f16, halves the gather bytes) -> agg[dst] (f32).
__global__ void __launch_bounds__(256)
scatter_h(const int* __restrict__ src, const int* __restrict__ dst,
          const _Float16* __restrict__ hPost, float* __restrict__ agg, int E) {
  int tid = blockIdx.x * 256 + threadIdx.x;
  int e = tid >> 5;
  if (e >= E) return;
  int s = src[e], d = dst[e];
  int f = (tid & 31) * 4;
  half4_t v = *(const half4_t*)(hPost + (size_t)s * HD + f);
  float* ap = agg + (size_t)d * HD + f;
  atomicAdd(ap + 0, (float)v[0]); atomicAdd(ap + 1, (float)v[1]);
  atomicAdd(ap + 2, (float)v[2]); atomicAdd(ap + 3, (float)v[3]);
}

// agg / max(cnt,1) -> f16
__global__ void __launch_bounds__(256)
agg_scale(const float* __restrict__ agg, const float* __restrict__ cnt,
          _Float16* __restrict__ out, int N) {
  int tid = blockIdx.x * 256 + threadIdx.x;
  int n = tid >> 5;
  if (n >= N) return;
  int f = (tid & 31) * 4;
  float r = 1.0f / fmaxf(cnt[n], 1.0f);
  float4 v = *(const float4*)(agg + (size_t)n * HD + f);
  half4_t o = { (_Float16)(v.x * r), (_Float16)(v.y * r),
                (_Float16)(v.z * r), (_Float16)(v.w * r) };
  *(half4_t*)(out + (size_t)n * HD + f) = o;
}

// relu + layernorm -> f16. One wave per node, 4 elems/lane (wave32).
__global__ void __launch_bounds__(256)
relu_ln(const float* __restrict__ Hbuf, const float* __restrict__ g,
        const float* __restrict__ b, _Float16* __restrict__ out, int N) {
  int n = blockIdx.x * 8 + (threadIdx.x >> 5);
  if (n >= N) return;
  int lane = threadIdx.x & 31;
  float4 v = *(const float4*)(Hbuf + (size_t)n * HD + lane * 4);
  v.x = fmaxf(v.x, 0.f); v.y = fmaxf(v.y, 0.f);
  v.z = fmaxf(v.z, 0.f); v.w = fmaxf(v.w, 0.f);
  float s = v.x + v.y + v.z + v.w;
  float s2 = v.x * v.x + v.y * v.y + v.z * v.z + v.w * v.w;
#pragma unroll
  for (int m = 16; m >= 1; m >>= 1) {
    s  += __shfl_xor(s,  m, 32);
    s2 += __shfl_xor(s2, m, 32);
  }
  float mean = s * (1.0f / HD);
  float var = s2 * (1.0f / HD) - mean * mean;
  float rs = rsqrtf(var + 1e-5f);
  int f = lane * 4;
  half4_t o = {
    (_Float16)((v.x - mean) * rs * g[f + 0] + b[f + 0]),
    (_Float16)((v.y - mean) * rs * g[f + 1] + b[f + 1]),
    (_Float16)((v.z - mean) * rs * g[f + 2] + b[f + 2]),
    (_Float16)((v.w - mean) * rs * g[f + 3] + b[f + 3]) };
  *(half4_t*)(out + (size_t)n * HD + f) = o;
}

// plain relu f32 -> f16 (final layer, no LN)
__global__ void __launch_bounds__(256)
relu_f16(const float* __restrict__ in, _Float16* __restrict__ out, int n4) {
  int i = blockIdx.x * 256 + threadIdx.x;   // over N*128/4
  if (i >= n4) return;
  float4 v = *(const float4*)(in + (size_t)i * 4);
  half4_t o = { (_Float16)fmaxf(v.x, 0.f), (_Float16)fmaxf(v.y, 0.f),
                (_Float16)fmaxf(v.z, 0.f), (_Float16)fmaxf(v.w, 0.f) };
  *(half4_t*)(out + (size_t)i * 4) = o;
}

// logits = mlpH @ w2 + b2 ; log_softmax over 5
__global__ void __launch_bounds__(256)
logits_kernel(const float* __restrict__ mlpH, const float* __restrict__ w2,
              const float* __restrict__ b2, float* __restrict__ out, int N) {
  int n = blockIdx.x * 256 + threadIdx.x;
  if (n >= N) return;
  const float* hr = mlpH + (size_t)n * HD;
  float acc[5];
#pragma unroll
  for (int j = 0; j < 5; ++j) acc[j] = b2[j];
  for (int k = 0; k < HD; ++k) {
    float hv = hr[k];
#pragma unroll
    for (int j = 0; j < 5; ++j) acc[j] += hv * w2[k * 5 + j];
  }
  float m = acc[0];
#pragma unroll
  for (int j = 1; j < 5; ++j) m = fmaxf(m, acc[j]);
  float sum = 0.f;
#pragma unroll
  for (int j = 0; j < 5; ++j) sum += expf(acc[j] - m);
  float lse = logf(sum);
#pragma unroll
  for (int j = 0; j < 5; ++j) out[(size_t)n * 5 + j] = acc[j] - m - lse;
}

// ---------------------------------------------------------------------------
extern "C" void kernel_launch(void* const* d_in, const int* in_sizes, int n_in,
                              void* d_out, int out_size, void* d_ws, size_t ws_size,
                              hipStream_t stream) {
  const float* x         = (const float*)d_in[0];
  const int*   ei        = (const int*)  d_in[1];
  const float* conv_w    = (const float*)d_in[2];
  const float* conv_b    = (const float*)d_in[3];
  const float* conv_sw   = (const float*)d_in[4];
  const float* conv_sb   = (const float*)d_in[5];
  const float* lin_w     = (const float*)d_in[6];
  const float* lin_b     = (const float*)d_in[7];
  const float* lin_sw    = (const float*)d_in[8];
  const float* lin_sb    = (const float*)d_in[9];
  const float* sage_lw   = (const float*)d_in[10];
  const float* sage_lb   = (const float*)d_in[11];
  const float* sage_rw   = (const float*)d_in[12];
  const float* ln_g      = (const float*)d_in[13];
  const float* ln_b      = (const float*)d_in[14];
  const float* mlp_w1    = (const float*)d_in[15];
  const float* mlp_b1    = (const float*)d_in[16];
  const float* mlp_w2    = (const float*)d_in[17];
  const float* mlp_b2    = (const float*)d_in[18];

  const int N = in_sizes[0] / 363;       // 100000 (divisible by 16)
  const int E = in_sizes[1] / 2;         // 1600000
  const int nTiles = N / 16;             // 6250
  const int* srcI = ei;
  const int* dstI = ei + E;

  // ---- workspace carve-up (~231 MB), with aliasing ----
  size_t off = 0;
  auto alloc = [&](size_t bytes) -> void* {
    void* p = (char*)d_ws + off;
    off += (bytes + 255) & ~(size_t)255;
    return p;
  };
  _Float16* cSelf = (_Float16*)alloc((size_t)N * KPAD * 2);   // later: xn (f32)
  _Float16* cNb   = (_Float16*)alloc((size_t)N * KPAD * 2);   // later: aggF16
  float*    hbuf  = (float*)   alloc((size_t)N * HD * 4);
  _Float16* hPost = (_Float16*)alloc((size_t)N * HD * 2);
  float*    agg   = (float*)   alloc((size_t)N * HD * 4);     // later: mlpH
  float*    cnt   = (float*)   alloc((size_t)N * 4);
  _Float16* linT  = (_Float16*)alloc(HD * KPAD * 2);
  _Float16* linST = (_Float16*)alloc(HD * KPAD * 2);
  _Float16* sqT[7];
  for (int i = 0; i < 7; ++i) sqT[i] = (_Float16*)alloc(HD * HD * 2);

  float*    xn     = (float*)cSelf;     // alias (cSelf dead after its GEMM)
  _Float16* aggF16 = cNb;               // alias (cNb dead after its GEMM)
  float*    mlpH   = agg;               // alias (agg dead before MLP)
  float*    emb    = (float*)d_out;                 // [N,128]
  float*    lsOut  = (float*)d_out + (size_t)N * HD; // [N,5]

  // ---- weight prep ----
  prep_padT<<<(HD * KPAD + 255) / 256, 256, 0, stream>>>(lin_w, lin_sw, linT, linST);
  const float* sq[7] = { sage_lw, sage_lw + HD * HD, sage_lw + 2 * HD * HD,
                         sage_rw, sage_rw + HD * HD, sage_rw + 2 * HD * HD,
                         mlp_w1 };
  for (int i = 0; i < 7; ++i)
    prep_T128<<<(HD * HD + 255) / 256, 256, 0, stream>>>(sq[i], sqT[i]);

  // ---- layer 0: convs -> f16 WMMA A-matrices; two 243->128 GEMMs ----
  conv_kernel<<<N, 256, 0, stream>>>(x, conv_w, conv_b, conv_sw, conv_sb, cNb, cSelf);
  int gGrid = (nTiles + 7) / 8;
  gemm_k256<<<gGrid, 256, 0, stream>>>(cSelf, linST, lin_sb, hbuf, nTiles); // self_x
  gemm_k256<<<gGrid, 256, 0, stream>>>(cNb,   linT,  lin_b,  xn,   nTiles); // xn

  // ---- degree counts (all edges) ----
  hipMemsetAsync(cnt, 0, (size_t)N * 4, stream);
  cnt_kernel<<<(E + 255) / 256, 256, 0, stream>>>(dstI, cnt, E);

  // ---- layer-0 scatter (self loops masked): h = self_x + sum(xn[src]) ----
  int sGrid = (E * 32 + 255) / 256;
  scatter_xn<<<sGrid, 256, 0, stream>>>(srcI, dstI, xn, hbuf, E);
  relu_ln<<<(N + 7) / 8, 256, 0, stream>>>(hbuf, ln_g, ln_b, hPost, N);

  // ---- 3 SAGE layers ----
  for (int j = 0; j < 3; ++j) {
    hipMemsetAsync(agg, 0, (size_t)N * HD * 4, stream);
    scatter_h<<<sGrid, 256, 0, stream>>>(srcI, dstI, hPost, agg, E);
    agg_scale<<<(N * 32 + 255) / 256, 256, 0, stream>>>(agg, cnt, aggF16, N);
    float* outBuf = (j == 2) ? emb : hbuf;   // last layer IS emb (pre-relu)
    gemm128<true><<<gGrid, 256, 0, stream>>>(aggF16, hPost, sqT[j], sqT[3 + j],
                                             sage_lb + (size_t)j * HD, outBuf, nTiles);
    if (j < 2)
      relu_ln<<<(N + 7) / 8, 256, 0, stream>>>(hbuf, ln_g + (size_t)(j + 1) * HD,
                                               ln_b + (size_t)(j + 1) * HD, hPost, N);
    else
      relu_f16<<<(N * 32 + 255) / 256, 256, 0, stream>>>(emb, hPost, N * 32);
  }

  // ---- post-MP MLP + log_softmax ----
  gemm128<false><<<gGrid, 256, 0, stream>>>(hPost, nullptr, sqT[6], nullptr,
                                            mlp_b1, mlpH, nTiles);
  logits_kernel<<<(N + 255) / 256, 256, 0, stream>>>(mlpH, mlp_w2, mlp_b2, lsOut, N);
  (void)n_in; (void)out_size; (void)ws_size;
}
